// BasicBlock_26671746908396
// MI455X (gfx1250) — matrix operand order repaired and checked by reading, not compiled, not run on hardware
//
#include <hip/hip_runtime.h>
#include <cstdint>
#include <cstddef>

// ---------------------------------------------------------------------------
// MI455X (gfx1250) L+S unrolled-recon step.
// Dominant cost (~0.45 TFLOP): 3x3x3 convs over (2,32,128,128,16) volumes as
// implicit GEMM on v_wmma_f32_16x16x32_bf16 (wave32).
//   M = positions (16-wide tile = full NT line), N = 32 cout (2 tiles),
//   K = 27 taps x 32 cin (one WMMA k-step per tap).
// Activations: bf16 channels-last with a zero halo [n][130][130][18][32]
// (38.9 MB -> L2-resident; halo removes ALL boundary predication from the
// WMMA inner loop). Each wave: 2 adjacent h-rows x 2 cout tiles = 4 accs,
// 108 WMMAs, B fragments reused from registers.
// Weights (54 KB) staged to LDS via the Tensor Data Mover.
// ---------------------------------------------------------------------------

typedef __attribute__((ext_vector_type(16))) __bf16 v16bf;
typedef __attribute__((ext_vector_type(8)))  float  v8f;
typedef unsigned int v4u __attribute__((ext_vector_type(4)));
typedef int          v4i __attribute__((ext_vector_type(4)));
typedef int          v8i __attribute__((ext_vector_type(8)));

struct Frag32 { uint4 lo, hi; };   // 32 bytes == one v16bf fragment

__device__ __forceinline__ unsigned short f2bf(float f) {
  __bf16 h = (__bf16)f;
  return __builtin_bit_cast(unsigned short, h);
}
__device__ __forceinline__ float bf2f(unsigned short u) {
  __bf16 h = __builtin_bit_cast(__bf16, u);
  return (float)h;
}
__device__ __forceinline__ float softplusf(float x) { return log1pf(expf(x)); }

#define NPOS 262144             // 128*128*16 logical positions per n-plane
// Padded (halo) activation layout: [n(2)][dp(130)][hp(130)][wp(18)][ci(32)]
#define DP 130
#define HP 130
#define WPD 18
#define PST_H (WPD * 32)            // 576
#define PST_D (HP * WPD * 32)       // 74880
#define PST_N (DP * HP * WPD * 32)  // 9734400
#define APAD_ELEMS (2 * PST_N)      // 19468800 bf16 elems (38.9 MB)
#define WP_ELEMS (2 * 27 * 32 * 16) // 27648 packed weight elems (54 KB)

__device__ __forceinline__ int padded_off(int n, int d, int h, int w, int ci) {
  return ((((n * DP) + d + 1) * HP + (h + 1)) * WPD + (w + 1)) * 32 + ci;
}

// ---------------------------------------------------------------------------
// TDM: stage the packed-weight block (13824 dwords, contiguous) into LDS.
// D# group0: count=1, lds_addr, global_addr, type=2. group1: data_size=4B,
// tensor_dim0 = tile_dim0 = 13824, tile_dim1 = 1, stride = 13824.
// Toolchain arity: therock headers present -> 6-arg builtin, else 5-arg.
// ---------------------------------------------------------------------------
#if __has_include(<hip/amd_detail/amd_gfx1250_TDM.h>)
#define ATHENA_TDM_6ARG 1
#endif

__device__ __forceinline__ void tdm_stage_weights(const unsigned short* wp,
                                                  unsigned short* wlds_p) {
  unsigned lds_off = (unsigned)(uintptr_t)wlds_p;   // LDS aperture: low 32 bits
  unsigned long long ga = (unsigned long long)(uintptr_t)wp;
  const unsigned ndw = WP_ELEMS / 2u;               // 13824 dwords

  v4u g0;
  g0.x = 1u;                                        // count=1, user descriptor
  g0.y = lds_off;                                   // lds_addr [63:32]
  g0.z = (unsigned)(ga & 0xffffffffu);              // global_addr lo
  g0.w = (unsigned)((ga >> 32) & 0x01ffffffu) | (2u << 30);  // addr hi | type=2

  v8i g1;
  g1[0] = (int)(2u << 16);                          // wg_mask=0, data_size=4B
  g1[1] = (int)((ndw & 0xffffu) << 16);             // tensor_dim0[15:0] @63:48
  g1[2] = (int)(((ndw >> 16) & 0xffffu) | (1u << 16)); // dim0 hi | dim1=1 lo
  g1[3] = (int)(ndw << 16);                         // dim1 hi=0 | tile_dim0
  g1[4] = 1;                                        // tile_dim1=1, tile_dim2=0
  g1[5] = (int)ndw;                                 // tensor_dim0_stride lo32
  g1[6] = 0;                                        // stride hi | dim1_stride
  g1[7] = 0;

  v4i gz = {0, 0, 0, 0};
#if defined(ATHENA_TDM_6ARG)
  v8i gz8 = {0, 0, 0, 0, 0, 0, 0, 0};
  __builtin_amdgcn_tensor_load_to_lds(g0, g1, gz, gz, gz8, 0);
#else
  __builtin_amdgcn_tensor_load_to_lds(g0, g1, gz, gz, 0);
#endif
  __builtin_amdgcn_s_wait_tensorcnt(0);
}

// ---------------------------------------------------------------------------
// Zero a workspace region (halo init).
// ---------------------------------------------------------------------------
__global__ void zero_ws_kernel(uint4* __restrict__ p, int n4) {
  int i = blockIdx.x * blockDim.x + threadIdx.x;
  if (i < n4) p[i] = make_uint4(0u, 0u, 0u, 0u);
}

// ---------------------------------------------------------------------------
// Pack f32 NCDHW weights (cout,cin,3,3,3) into per-lane bf16 B fragments:
// wp[tile(2)][tap(27)][lane(32)][elem(16)], zero-padded to 32x32 channels.
// Lane L holds N=(L&15)+16*tile, K = (elem&7) + 16*(elem>>3) + (L<16?0:8).
// ---------------------------------------------------------------------------
__global__ void pack_weights_kernel(const float* __restrict__ wsrc,
                                    unsigned short* __restrict__ wp,
                                    int cout, int cin) {
  int idx = blockIdx.x * blockDim.x + threadIdx.x;   // WP_ELEMS threads
  if (idx >= WP_ELEMS) return;
  int e    = idx & 15;
  int lane = (idx >> 4) & 31;
  int tg   = idx >> 9;
  int tap  = tg % 27;
  int tile = tg / 27;
  int co = (lane & 15) + 16 * tile;
  int ci = (e & 7) + ((e >> 3) << 4) + ((lane < 16) ? 0 : 8);
  float v = 0.f;
  if (co < cout && ci < cin)
    v = wsrc[(co * cin + ci) * 27 + tap];
  wp[idx] = f2bf(v);
}

// ---------------------------------------------------------------------------
// f32 NCDHW (2,cin,128,128,16) -> halo-padded bf16 channels-last (interior).
// ---------------------------------------------------------------------------
__global__ void pack_acts_kernel(const float* __restrict__ src,
                                 unsigned short* __restrict__ dst, int cin) {
  int idx = blockIdx.x * blockDim.x + threadIdx.x;   // 2*NPOS*32 threads
  int ci  = idx & 31;
  int pos = idx >> 5;
  int w = pos & 15;
  int h = (pos >> 4) & 127;
  int d = (pos >> 11) & 127;
  int n = pos >> 18;
  float v = 0.f;
  if (ci < cin)
    v = src[(((n * cin + ci) * 128 + d) * 128 + h) * 16 + w];
  dst[padded_off(n, d, h, w, ci)] = f2bf(v);
}

// ---------------------------------------------------------------------------
// Branch-free implicit-GEMM 3x3x3 conv on halo-padded input.
// Block = 256 threads = 8 waves; wave owns 2 adjacent h-rows x 32 cout.
// ---------------------------------------------------------------------------
__global__ __launch_bounds__(256, 2)
void conv3d_wmma_kernel(const unsigned short* __restrict__ apack,
                        const unsigned short* __restrict__ wp,
                        unsigned short* __restrict__ opack,
                        int relu) {
  __shared__ __align__(32) unsigned short wlds[WP_ELEMS];  // 55296 B

  if (threadIdx.x < 32) tdm_stage_weights(wp, wlds);       // wave 0 DMA
  __syncthreads();

  int tid  = threadIdx.x;
  int lane = tid & 31;
  int wvid = tid >> 5;
  int pair = blockIdx.x * 8 + wvid;     // 0..16383 == 2*128*64 row-pairs
  int hh = (pair & 63) * 2;             // even h base
  int d  = (pair >> 6) & 127;
  int n  = pair >> 13;

  int c0   = (lane < 16) ? 0 : 8;       // ISA 16-bit A layout channel base
  int wl15 = lane & 15;                 // M (= w position)

  v8f acc00 = {}, acc01 = {}, acc10 = {}, acc11 = {};

  const unsigned short* base = apack + (size_t)n * PST_N + c0;

  for (int t = 0; t < 27; ++t) {
    int kd = t / 9, kh = (t / 3) % 3, kw = t % 3;
    const unsigned short* a0p =
        base + (d + kd) * PST_D + (hh + kh) * PST_H + (wl15 + kw) * 32;
    const unsigned short* a1p = a0p + PST_H;   // next h-row
    if (kh == 0 && kw == 0)
      __builtin_prefetch((const void*)(a0p + PST_D), 0, 1);

    Frag32 f0, f1;
    f0.lo = *(const uint4*)a0p;
    f0.hi = *(const uint4*)(a0p + 16);
    f1.lo = *(const uint4*)a1p;
    f1.hi = *(const uint4*)(a1p + 16);
    v16bf a0 = __builtin_bit_cast(v16bf, f0);
    v16bf a1 = __builtin_bit_cast(v16bf, f1);
    v16bf b0 = *(const v16bf*)(&wlds[(t * 32 + lane) * 16]);
    v16bf b1 = *(const v16bf*)(&wlds[((27 + t) * 32 + lane) * 16]);

    acc00 = __builtin_amdgcn_wmma_f32_16x16x32_bf16(false, a0, false, b0,
                                                    (short)0, acc00, false, false);
    acc01 = __builtin_amdgcn_wmma_f32_16x16x32_bf16(false, a0, false, b1,
                                                    (short)0, acc01, false, false);
    acc10 = __builtin_amdgcn_wmma_f32_16x16x32_bf16(false, a1, false, b0,
                                                    (short)0, acc10, false, false);
    acc11 = __builtin_amdgcn_wmma_f32_16x16x32_bf16(false, a1, false, b1,
                                                    (short)0, acc11, false, false);
  }

  // D layout: VGPR r, lanes<16 -> M=r, lanes>=16 -> M=8+r; N = lane&15.
  int co   = lane & 15;
  int wofs = (lane >= 16) ? 8 : 0;
  int ob0  = padded_off(n, d, hh, 0, 0) - 32;      // w=-? base, add (w+1)*32
  int ob1  = ob0 + PST_H;
#pragma unroll
  for (int r = 0; r < 8; ++r) {
    int w = r + wofs;
    float v00 = acc00[r], v01 = acc01[r], v10 = acc10[r], v11 = acc11[r];
    if (relu) {
      v00 = fmaxf(v00, 0.f); v01 = fmaxf(v01, 0.f);
      v10 = fmaxf(v10, 0.f); v11 = fmaxf(v11, 0.f);
    }
    opack[ob0 + (w + 1) * 32 + co]      = f2bf(v00);
    opack[ob0 + (w + 1) * 32 + co + 16] = f2bf(v01);
    opack[ob1 + (w + 1) * 32 + co]      = f2bf(v10);
    opack[ob1 + (w + 1) * 32 + co + 16] = f2bf(v11);
  }
}

// ---------------------------------------------------------------------------
// Derived scalars: gamma, c, lamL, lamS, lspL, lspS.
// ---------------------------------------------------------------------------
__global__ void scalars_kernel(const float* gp, const float* lsp_step,
                               const float* lamL, const float* lamS,
                               const float* lspL, const float* lspS,
                               float* out) {
  if (blockIdx.x == 0 && threadIdx.x == 0) {
    float g = softplusf(gp[0]) + 1e-6f;
    out[0] = g;
    out[1] = (softplusf(lsp_step[0]) + 1e-6f) / g;
    out[2] = softplusf(lamL[0]) + 1e-8f;
    out[3] = softplusf(lamS[0]) + 1e-8f;
    out[4] = softplusf(lspL[0]) + 1e-8f;
    out[5] = softplusf(lspS[0]) + 1e-8f;
  }
}

// y[n][pos] = base - gamma*pt - gamma*pb (pb = ch 0 of bwd-CNN output, padded)
__global__ void combine_y_kernel(const float* __restrict__ Br,
                                 const float* __restrict__ Bi,
                                 const float* __restrict__ ptr_,
                                 const float* __restrict__ pti,
                                 const unsigned short* __restrict__ pb,
                                 const float* __restrict__ scal,
                                 float* __restrict__ y) {
  int idx = blockIdx.x * blockDim.x + threadIdx.x;   // 2*NPOS
  int n   = idx >> 18;
  int pos = idx & (NPOS - 1);
  int w = pos & 15;
  int h = (pos >> 4) & 127;
  int d = pos >> 11;
  float g  = scal[0];
  float pv = bf2f(pb[padded_off(n, d, h, w, 0)]);
  float b  = n ? Bi[pos] : Br[pos];
  float pt = n ? pti[pos] : ptr_[pos];
  y[idx] = b - g * pt - g * pv;
}

// FiLM params: p = emb @ W^T + b ; fp[j] = 0.1*tanh(p[j]), j in [0,64)
__global__ void film_params_kernel(const float* __restrict__ W,
                                   const float* __restrict__ b,
                                   const float* __restrict__ emb,
                                   float* __restrict__ fp) {
  int j = threadIdx.x;
  if (j < 64) {
    float acc = b[j];
    for (int k = 0; k < 128; ++k) acc += emb[k] * W[j * 128 + k];
    fp[j] = 0.1f * tanhf(acc);
  }
}

// x = relu(x*(1+scale[c]) + bias[c]) in-place on padded tensor (interior).
__global__ void film_apply_kernel(unsigned short* __restrict__ x,
                                  const float* __restrict__ fp) {
  int idx = blockIdx.x * blockDim.x + threadIdx.x;   // 2*NPOS*32
  int ci  = idx & 31;
  int pos = idx >> 5;
  int w = pos & 15;
  int h = (pos >> 4) & 127;
  int d = (pos >> 11) & 127;
  int n = pos >> 18;
  int po = padded_off(n, d, h, w, ci);
  float v = bf2f(x[po]);
  v = v * (1.f + fp[ci]) + fp[32 + ci];
  x[po] = f2bf(fmaxf(v, 0.f));
}

// p' = proj_inf(c * (t_conv + p_in)) -> padded bf16 interior (next CNN input).
__global__ void proj_update_p_kernel(const unsigned short* __restrict__ t_opack,
                                     const float* __restrict__ p_in,
                                     const float* __restrict__ scal, int lsp_idx,
                                     unsigned short* __restrict__ p_out) {
  int idx = blockIdx.x * blockDim.x + threadIdx.x;   // NPOS*32
  int ci  = idx & 31;
  int pos = idx >> 5;
  int w = pos & 15;
  int h = (pos >> 4) & 127;
  int d = pos >> 11;
  int po0 = padded_off(0, d, h, w, ci);
  int po1 = padded_off(1, d, h, w, ci);
  float cf  = scal[1];
  float lam = scal[lsp_idx];
  float tr = bf2f(t_opack[po0]) + p_in[ci * NPOS + pos];
  float ti = bf2f(t_opack[po1]) + p_in[(32 + ci) * NPOS + pos];
  tr *= cf; ti *= cf;
  float mag = sqrtf(tr * tr + ti * ti) + 1e-12f;
  float f = fminf(1.f, lam / mag);
  p_out[po0] = f2bf(tr * f);
  p_out[po1] = f2bf(ti * f);
}

// out = base - gamma*pt - gamma*pb' (both planes of one branch)
__global__ void final_update_kernel(const float* __restrict__ Br,
                                    const float* __restrict__ Bi,
                                    const float* __restrict__ ptr_,
                                    const float* __restrict__ pti,
                                    const unsigned short* __restrict__ pb,
                                    const float* __restrict__ scal,
                                    float* __restrict__ outr,
                                    float* __restrict__ outi) {
  int pos = blockIdx.x * blockDim.x + threadIdx.x;   // NPOS
  int w = pos & 15;
  int h = (pos >> 4) & 127;
  int d = pos >> 11;
  float g = scal[0];
  float pbr = bf2f(pb[padded_off(0, d, h, w, 0)]);
  float pbi = bf2f(pb[padded_off(1, d, h, w, 0)]);
  outr[pos] = Br[pos] - g * ptr_[pos] - g * pbr;
  outi[pos] = Bi[pos] - g * pti[pos] - g * pbi;
}

// ---------------------------------------------------------------------------
extern "C" void kernel_launch(void* const* d_in, const int* in_sizes, int n_in,
                              void* d_out, int out_size, void* d_ws, size_t ws_size,
                              hipStream_t stream) {
  (void)in_sizes; (void)n_in; (void)out_size; (void)ws_size;

  const float* Lr   = (const float*)d_in[4];
  const float* Li   = (const float*)d_in[5];
  const float* Sr   = (const float*)d_in[6];
  const float* Si   = (const float*)d_in[7];
  const float* ptLr = (const float*)d_in[8];
  const float* ptLi = (const float*)d_in[9];
  const float* ptSr = (const float*)d_in[10];
  const float* ptSi = (const float*)d_in[11];
  const float* pL   = (const float*)d_in[12];
  const float* pS   = (const float*)d_in[13];
  const float* emb  = (const float*)d_in[16];
  const float* lamL = (const float*)d_in[17];
  const float* lamS = (const float*)d_in[18];
  const float* lspL = (const float*)d_in[19];
  const float* lspS = (const float*)d_in[20];
  const float* gam  = (const float*)d_in[21];
  const float* lst  = (const float*)d_in[22];
  const float* WfL  = (const float*)d_in[23];
  const float* bfL  = (const float*)d_in[24];
  const float* WfS  = (const float*)d_in[25];
  const float* bfS  = (const float*)d_in[26];
  const float* c1f_l = (const float*)d_in[27];
  const float* c2f_l = (const float*)d_in[28];
  const float* c3f_l = (const float*)d_in[29];
  const float* c1b_l = (const float*)d_in[30];
  const float* c2b_l = (const float*)d_in[31];
  const float* c3b_l = (const float*)d_in[32];
  const float* c1f_s = (const float*)d_in[33];
  const float* c2f_s = (const float*)d_in[34];
  const float* c3f_s = (const float*)d_in[35];
  const float* c1b_s = (const float*)d_in[36];
  const float* c2b_s = (const float*)d_in[37];
  const float* c3b_s = (const float*)d_in[38];

  // Workspace carve-up (all 16B-aligned regions).
  char* ws = (char*)d_ws;
  const size_t APAD_BYTES = (size_t)APAD_ELEMS * 2;      // 38.9 MB
  unsigned short* apA = (unsigned short*)(ws);
  unsigned short* apB = (unsigned short*)(ws + APAD_BYTES);
  float* ybuf         = (float*)(ws + 2 * APAD_BYTES);   // 2*NPOS f32
  unsigned short* wp  = (unsigned short*)(ws + 2 * APAD_BYTES + (size_t)2 * NPOS * 4);
  float* fp           = (float*)(ws + 2 * APAD_BYTES + (size_t)2 * NPOS * 4 + 65536);
  float* scal         = fp + 64;
  float* out          = (float*)d_out;

  // Zero the halo-padded buffers once per call (halos stay zero thereafter).
  const int N4 = (int)(APAD_BYTES / 16);                 // 2433600 uint4
  zero_ws_kernel<<<(N4 + 255) / 256, 256, 0, stream>>>((uint4*)apA, N4);
  zero_ws_kernel<<<(N4 + 255) / 256, 256, 0, stream>>>((uint4*)apB, N4);

  scalars_kernel<<<1, 32, 0, stream>>>(gam, lst, lamL, lamS, lspL, lspS, scal);

  auto conv = [&](const float* w, int cout, int cin,
                  const unsigned short* in, unsigned short* outp, int relu) {
    pack_weights_kernel<<<108, 256, 0, stream>>>(w, wp, cout, cin);
    conv3d_wmma_kernel<<<2048, 256, 0, stream>>>(in, wp, outp, relu);
  };

  auto branch = [&](const float* Br, const float* Bi,
                    const float* ptr_, const float* pti,
                    const float* pin, const float* Wf, const float* bfv,
                    const float* w1f, const float* w2f, const float* w3f,
                    const float* w1b, const float* w2b, const float* w3b,
                    int lsp_idx, float* outr, float* outi) {
    // pb = bwd_cnn(p)
    pack_acts_kernel<<<65536, 256, 0, stream>>>(pin, apA, 32);
    conv(w1b, 32, 32, apA, apB, 1);
    conv(w2b, 32, 32, apB, apA, 1);
    conv(w3b, 1, 32, apA, apB, 0);
    // y = base - g*pt - g*pb
    combine_y_kernel<<<2048, 256, 0, stream>>>(Br, Bi, ptr_, pti, apB, scal, ybuf);
    // forward CNN on y (cin=1 padded)
    pack_acts_kernel<<<65536, 256, 0, stream>>>(ybuf, apA, 1);
    conv(w1f, 32, 1, apA, apB, 1);
    conv(w2f, 32, 32, apB, apA, 0);
    film_params_kernel<<<1, 64, 0, stream>>>(Wf, bfv, emb, fp);
    film_apply_kernel<<<65536, 256, 0, stream>>>(apA, fp);
    conv(w3f, 32, 32, apA, apB, 0);
    // p' = proj_inf(c*(t + p)) -> padded bf16 for next bwd CNN
    proj_update_p_kernel<<<32768, 256, 0, stream>>>(apB, pin, scal, lsp_idx, apA);
    // pb' = bwd_cnn(p')
    conv(w1b, 32, 32, apA, apB, 1);
    conv(w2b, 32, 32, apB, apA, 1);
    conv(w3b, 1, 32, apA, apB, 0);
    final_update_kernel<<<1024, 256, 0, stream>>>(Br, Bi, ptr_, pti, apB, scal,
                                                  outr, outi);
  };

  branch(Lr, Li, ptLr, ptLi, pL, WfL, bfL,
         c1f_l, c2f_l, c3f_l, c1b_l, c2b_l, c3b_l,
         4, out + 0 * NPOS, out + 1 * NPOS);
  branch(Sr, Si, ptSr, ptSi, pS, WfS, bfS,
         c1f_s, c2f_s, c3f_s, c1b_s, c2b_s, c3b_s,
         5, out + 2 * NPOS, out + 3 * NPOS);
}
